// FullAttention_4595615007078
// MI455X (gfx1250) — matrix operand order repaired and checked by reading, not compiled, not run on hardware
//
#include <hip/hip_runtime.h>
#include <math.h>

typedef __attribute__((ext_vector_type(16))) _Float16 v16h;
typedef __attribute__((ext_vector_type(8)))  _Float16 v8h;
typedef __attribute__((ext_vector_type(8)))  float    v8f;

#define BATCH 2
#define SEQ   2048
#define HEADS 16
#define EDIM  64
#define BM    64     // query rows per workgroup
#define BN    64     // key rows per iteration
#define NW    4      // waves per workgroup (wave32)
#define PADH  72     // halves per LDS row: multiple of 8 -> 16B-aligned b128 reads

static __device__ __forceinline__ v16h cat16(v8h lo, v8h hi) {
  return __builtin_shufflevector(lo, hi, 0,1,2,3,4,5,6,7,8,9,10,11,12,13,14,15);
}

// packed f32x2 -> f16x2 conversion (v_cvt_pk_rtz_f16_f32)
static __device__ __forceinline__ unsigned pkh2(float a, float b) {
#if __has_builtin(__builtin_amdgcn_cvt_pkrtz)
  return __builtin_bit_cast(unsigned, __builtin_amdgcn_cvt_pkrtz(a, b));
#else
  union { _Float16 h[2]; unsigned u; } x;
  x.h[0] = (_Float16)a; x.h[1] = (_Float16)b; return x.u;
#endif
}

// xor-shuffle within each 16-lane group via v_permlane16_b32 (pure VALU)
static __device__ __forceinline__ float plx(float x, unsigned s0, unsigned s1, int step) {
#if __has_builtin(__builtin_amdgcn_permlane16)
  int xi = __builtin_bit_cast(int, x);
  int r  = __builtin_amdgcn_permlane16(xi, xi, (int)s0, (int)s1, false, false);
  (void)step;
  return __builtin_bit_cast(float, r);
#else
  (void)s0; (void)s1;
  return __shfl_xor(x, step, 32);
#endif
}

static __device__ __forceinline__ float fexp2(float x) {
#if __has_builtin(__builtin_amdgcn_exp2f)
  return __builtin_amdgcn_exp2f(x);
#else
  return exp2f(x);
#endif
}

__global__ __launch_bounds__(NW * 32)
void fa_fwd_causal(const float* __restrict__ q,
                   const float* __restrict__ k,
                   const float* __restrict__ v,
                   float* __restrict__ out)
{
  __shared__ __align__(16) _Float16 ldsK[BN * PADH];          // K tile  [n][e]
  __shared__ __align__(16) _Float16 ldsV[EDIM * PADH];        // V tile transposed [e][n]
  __shared__ __align__(16) _Float16 ldsP[NW * 16 * PADH];     // per-wave P rows [m][n]

  const int tid  = threadIdx.x;
  const int lane = tid & 31;
  const int wave = tid >> 5;
  const int l16  = lane & 15;
  const int hi   = (lane >> 4) & 1;   // which 16-lane half of the wave
  const int qblk = blockIdx.x;
  const int h    = blockIdx.y;
  const int b    = blockIdx.z;
  const int q0   = qblk * BM;

  const float kScale = 0.18033688011112042f;  // log2(e)/sqrt(E), E = 64

  // ---- load this wave's 16 Q rows into WMMA-A register layout, pre-scaled
  // so scores are already in exp2 units: s' = (q*log2e/8) . k
  const int    myrow = q0 + wave * 16 + l16;
  const float* qp    = q + (((size_t)b * SEQ + myrow) * HEADS + h) * EDIM;
  const int    base8 = hi ? 8 : 0;
  v16h qa[2];
  #pragma unroll
  for (int c = 0; c < 2; ++c) {
    const int e0 = c * 32 + base8;
    #pragma unroll
    for (int i = 0; i < 8; ++i) {
      qa[c][i]     = (_Float16)(qp[e0 + i]      * kScale);
      qa[c][i + 8] = (_Float16)(qp[e0 + 16 + i] * kScale);
    }
  }

  // all-ones B matrix for WMMA row-sum of P
  v16h ones;
  #pragma unroll
  for (int i = 0; i < 16; ++i) ones[i] = (_Float16)1.0f;

  float m_run[8];
  v8f   acc[4], lsum;
  #pragma unroll
  for (int r = 0; r < 8; ++r) m_run[r] = -INFINITY;
  #pragma unroll
  for (int t = 0; t < 4; ++t) acc[t] = (v8f){0.f,0.f,0.f,0.f,0.f,0.f,0.f,0.f};
  lsum = (v8f){0.f,0.f,0.f,0.f,0.f,0.f,0.f,0.f};

  for (int kb = 0; kb <= qblk; ++kb) {
    const int n0 = kb * BN;
    __syncthreads();   // all waves done reading previous K/V tile
    // ---- cooperative K/V tile load: f32 global -> f16 LDS (V stored transposed)
    // each thread handles two adjacent key rows so f16 stores are packed
    #pragma unroll
    for (int it = 0; it < 4; ++it) {
      const int idx = it * (NW * 32) + tid;   // 0..511
      const int n   = (idx >> 4) * 2;         // 0,2,...,62
      const int e4  = (idx & 15) << 2;        // 0,4,...,60
      const size_t g0 = (((size_t)b * SEQ + (n0 + n)) * HEADS + h) * EDIM + e4;
      const size_t g1 = g0 + (size_t)HEADS * EDIM;
      const float4 k0 = *(const float4*)(k + g0);
      const float4 k1 = *(const float4*)(k + g1);
      const float4 v0 = *(const float4*)(v + g0);
      const float4 v1 = *(const float4*)(v + g1);
      // K rows [n][e4..e4+3]
      *(uint2*)&ldsK[(n + 0) * PADH + e4] = make_uint2(pkh2(k0.x, k0.y), pkh2(k0.z, k0.w));
      *(uint2*)&ldsK[(n + 1) * PADH + e4] = make_uint2(pkh2(k1.x, k1.y), pkh2(k1.z, k1.w));
      // V transposed [e][n], rows n/n+1 packed per b32
      *(unsigned*)&ldsV[(e4 + 0) * PADH + n] = pkh2(v0.x, v1.x);
      *(unsigned*)&ldsV[(e4 + 1) * PADH + n] = pkh2(v0.y, v1.y);
      *(unsigned*)&ldsV[(e4 + 2) * PADH + n] = pkh2(v0.z, v1.z);
      *(unsigned*)&ldsV[(e4 + 3) * PADH + n] = pkh2(v0.w, v1.w);
      // prefetch next key block into cache while we compute on this one
      if (kb < qblk) {
        __builtin_prefetch(k + g0 + (size_t)BN * HEADS * EDIM, 0, 1);
        __builtin_prefetch(v + g0 + (size_t)BN * HEADS * EDIM, 0, 1);
      }
    }
    __syncthreads();

    // ---- S = Q K^T : 4 key-tiles x (contraction E=64 -> 2 WMMA)
    v8f s[4];
    #pragma unroll
    for (int t = 0; t < 4; ++t) {
      v8f cacc = (v8f){0.f,0.f,0.f,0.f,0.f,0.f,0.f,0.f};
      #pragma unroll
      for (int c = 0; c < 2; ++c) {
        // B (32x16 f16): lane col N = l16; K = c*32 + hi*8 + [0..8) and c*32 + 16 + hi*8 + [0..8)
        const v8h* bp = (const v8h*)&ldsK[(t * 16 + l16) * PADH + c * 32 + hi * 8];
        v16h bm = cat16(bp[0], bp[2]);
        cacc = __builtin_amdgcn_wmma_f32_16x16x32_f16(false, qa[c], false, bm,
                                                      (short)0, cacc, false, false);
      }
      s[t] = cacc;
    }

    // ---- causal mask (diagonal block only); C layout: M = r + 8*hi, N = l16
    if (kb == qblk) {
      #pragma unroll
      for (int t = 0; t < 4; ++t) {
        const int colc = t * 16 + l16;
        #pragma unroll
        for (int r = 0; r < 8; ++r) {
          const int rowc = wave * 16 + r + hi * 8;
          if (colc > rowc) s[t][r] = -INFINITY;
        }
      }
    }

    // ---- online softmax: each 16-lane group owns its 8 rows end-to-end
    float rmax[8];
    #pragma unroll
    for (int r = 0; r < 8; ++r)
      rmax[r] = fmaxf(fmaxf(s[0][r], s[1][r]), fmaxf(s[2][r], s[3][r]));
    #pragma unroll
    for (int r = 0; r < 8; ++r) {
      rmax[r] = fmaxf(rmax[r], plx(rmax[r], 0xFEDCBA98u, 0x76543210u, 8));
      rmax[r] = fmaxf(rmax[r], plx(rmax[r], 0x32107654u, 0xBA98FEDCu, 4));
      rmax[r] = fmaxf(rmax[r], plx(rmax[r], 0x54761032u, 0xDCFE98BAu, 2));
      rmax[r] = fmaxf(rmax[r], plx(rmax[r], 0x67452301u, 0xEFCDAB89u, 1));
    }

    float fac[8];
    #pragma unroll
    for (int r = 0; r < 8; ++r) {
      const float mn = fmaxf(m_run[r], rmax[r]);
      fac[r]   = fexp2(m_run[r] - mn);   // scores already in exp2 units
      m_run[r] = mn;
      lsum[r] *= fac[r];
    }
    #pragma unroll
    for (int t = 0; t < 4; ++t) {
      #pragma unroll
      for (int r = 0; r < 8; ++r) acc[t][r] *= fac[r];
    }

    #pragma unroll
    for (int t = 0; t < 4; ++t) {
      #pragma unroll
      for (int r = 0; r < 8; ++r) {
        const float p = fexp2(s[t][r] - m_run[r]);
        ldsP[(wave * 16 + r + hi * 8) * PADH + t * 16 + l16] = (_Float16)p;
      }
    }

    // same-wave DS store -> load ordering for the P strip
    asm volatile("s_wait_dscnt 0" ::: "memory");

    // ---- O += P V and row-sums lsum += P . 1 (contraction: 2 chunks of 32)
    #pragma unroll
    for (int c = 0; c < 2; ++c) {
      const v8h* ap = (const v8h*)&ldsP[(wave * 16 + l16) * PADH + c * 32 + base8];
      v16h am = cat16(ap[0], ap[2]);   // halves [0..8) and [16..24): A-layout chunks
      #pragma unroll
      for (int t = 0; t < 4; ++t) {
        const v8h* vp = (const v8h*)&ldsV[(t * 16 + l16) * PADH + c * 32 + base8];
        v16h bm = cat16(vp[0], vp[2]);
        acc[t] = __builtin_amdgcn_wmma_f32_16x16x32_f16(false, am, false, bm,
                                                        (short)0, acc[t], false, false);
      }
      lsum = __builtin_amdgcn_wmma_f32_16x16x32_f16(false, am, false, ones,
                                                    (short)0, lsum, false, false);
    }
  }

  // ---- normalize rows (rcp + mul) and store fp32 output [B,L,H,E]
  float inv[8];
  #pragma unroll
  for (int r = 0; r < 8; ++r) {
#if __has_builtin(__builtin_amdgcn_rcpf)
    inv[r] = __builtin_amdgcn_rcpf(lsum[r]);
#else
    inv[r] = 1.0f / lsum[r];
#endif
  }
  #pragma unroll
  for (int t = 0; t < 4; ++t) {
    #pragma unroll
    for (int r = 0; r < 8; ++r) {
      const int row = q0 + wave * 16 + r + hi * 8;
      const int e   = t * 16 + l16;
      out[(((size_t)b * SEQ + row) * HEADS + h) * EDIM + e] = acc[t][r] * inv[r];
    }
  }
}

extern "C" void kernel_launch(void* const* d_in, const int* in_sizes, int n_in,
                              void* d_out, int out_size, void* d_ws, size_t ws_size,
                              hipStream_t stream) {
  (void)in_sizes; (void)n_in; (void)out_size; (void)d_ws; (void)ws_size;
  const float* q = (const float*)d_in[0];
  const float* k = (const float*)d_in[1];
  const float* v = (const float*)d_in[2];
  float* o = (float*)d_out;
  dim3 grid(SEQ / BM, HEADS, BATCH);
  fa_fwd_causal<<<grid, NW * 32, 0, stream>>>(q, k, v, o);
}
